// Attention_54623394071059
// MI455X (gfx1250) — matrix operand order repaired and checked by reading, not compiled, not run on hardware
//
#include <hip/hip_runtime.h>
#include <hip/hip_bf16.h>
#include <cstdint>

// ---------------------------------------------------------------------------
// Causal MHA for MI455X (gfx1250, wave32, v_wmma_f32_16x16x32_f16).
//  - All WMMA operands pre-converted to f16 once (x, Wq/k/v, Wp); f32 accum.
//  - Transposed-flash: S^T = K Q^T -> per-lane softmax stats, P^T feeds the
//    next WMMA B operand with only xor-16 shuffles.
//  - K/V tiles staged in LDS via async global->LDS loads, shared by 4 waves.
//  - All GEMM orientations chosen so epilogue stores are contiguous b128.
// ---------------------------------------------------------------------------

constexpr int B_ = 2;
constexpr int T_ = 2048;
constexpr int E_ = 1024;
constexpr int H_ = 16;
constexpr int D_ = 64;
constexpr float SCALE_ = 0.125f;   // D^-0.5

constexpr int KSTR = 72;   // padded LDS row stride (f16) for 32x64 K tile
constexpr int VSTR = 40;   // padded LDS row stride (f16) for 64x32 V^T tile

typedef __attribute__((ext_vector_type(16))) _Float16 v16h;
typedef __attribute__((ext_vector_type(8)))  _Float16 v8h;
typedef __attribute__((ext_vector_type(8)))  float    v8f;
typedef __attribute__((ext_vector_type(4)))  float    v4f;

// A-fragment (16x32 f16, lane = M row): element e -> K index
__device__ __forceinline__ int kA(int lane, int e) {
    return ((lane & 16) ? 16 : 0) + e;
}
// B-fragment (32x16 f16, lane = N col): element e -> K index
__device__ __forceinline__ int kB(int lane, int e) {
    return ((lane & 16) ? 16 : 0) + e;
}

template <bool IS_A, typename SRC>
__device__ __forceinline__ v16h load_frag(const SRC* __restrict__ base,
                                          int rowStride, int lane, int kbase) {
    const SRC* p = base + (size_t)(lane & 15) * rowStride + kbase;
    v16h f;
#pragma unroll
    for (int e = 0; e < 16; ++e) {
        int k = IS_A ? kA(lane, e) : kB(lane, e);
        f[e] = (_Float16)p[k];
    }
    return f;
}

__device__ __forceinline__ v8f wmma_f16(v16h a, v16h b, v8f c) {
    return __builtin_amdgcn_wmma_f32_16x16x32_f16(false, a, false, b,
                                                  (short)0, c, false, false);
}

// CDNA5 async global->LDS 16B copy (tracked by ASYNCcnt).
__device__ __forceinline__ void async_copy16(uint32_t ldsOff, const void* gaddr) {
    uint64_t ga = (uint64_t)(uintptr_t)gaddr;
    asm volatile("global_load_async_to_lds_b128 %0, %1, off"
                 :: "v"(ldsOff), "v"(ga) : "memory");
}
__device__ __forceinline__ void wait_async0() {
    asm volatile("s_wait_asynccnt 0" ::: "memory");
}

// ---------------------------------------------------------------------------
__global__ void cvt_f16_kernel(const float* __restrict__ src,
                               _Float16* __restrict__ dst, int n) {
    int i = blockIdx.x * blockDim.x + threadIdx.x;
    if (i < n) dst[i] = (_Float16)src[i];
}

// ---------------------------------------------------------------------------
// QKV per-head diagonal-block projection (all-f16 operands).
// One wave per (b,h,16-row t tile).
// q,k computed as q^T = Wh * X^T (lane = t) -> each lane stores 8 contiguous
// f16 of one q/k row (b128). v computed as X * Wv^T (lane = d) -> each lane
// stores 8 contiguous f16 of one v^T row (b128).
// ---------------------------------------------------------------------------
__global__ void qkv_proj_kernel(const _Float16* __restrict__ xH,
                                const _Float16* __restrict__ WqH,
                                const float* __restrict__ bq,
                                const _Float16* __restrict__ WkH,
                                const float* __restrict__ bk,
                                const _Float16* __restrict__ WvH,
                                const float* __restrict__ bv,
                                _Float16* __restrict__ qws,
                                _Float16* __restrict__ kws,
                                _Float16* __restrict__ vTws) {
    int gtid = blockIdx.x * blockDim.x + threadIdx.x;
    int wid  = gtid >> 5;
    int lane = gtid & 31;
    const int TT = T_ / 16;
    int tTile = wid % TT;
    int bh    = wid / TT;
    if (bh >= B_ * H_) return;
    int h = bh % H_;
    int tBase = tTile * 16;
    int hi = (lane >> 4) & 1;
    int nn = lane & 15;

    const _Float16* xbase = xH + ((size_t)(bh / H_) * T_ + tBase) * E_ + h * D_;
    // X^T as B-fragments (lane = t column) for q/k; X as A-fragments for v.
    v16h xb0 = load_frag<false>(xbase, E_, lane, 0);
    v16h xb1 = load_frag<false>(xbase, E_, lane, 32);
    v16h xa0 = load_frag<true >(xbase, E_, lane, 0);
    v16h xa1 = load_frag<true >(xbase, E_, lane, 32);

    // ---- Q and K: transposed orientation, contiguous row stores ----
#pragma unroll
    for (int pr = 0; pr < 2; ++pr) {
        const _Float16* W    = pr ? WkH : WqH;
        const float*    bias = pr ? bk : bq;
        _Float16* dstRow = (pr ? kws : qws) +
                           ((size_t)bh * T_ + tBase + nn) * D_;
#pragma unroll
        for (int j = 0; j < 4; ++j) {
            const _Float16* wrow = W + (size_t)(h * D_ + j * 16) * D_;
            v16h wa0 = load_frag<true>(wrow, D_, lane, 0);
            v16h wa1 = load_frag<true>(wrow, D_, lane, 32);
            v8f acc = {};
            acc = wmma_f16(wa0, xb0, acc);
            acc = wmma_f16(wa1, xb1, acc);
            const float* bptr = bias + h * D_ + j * 16 + 8 * hi;
            v8h hv;
#pragma unroll
            for (int r = 0; r < 8; ++r) hv[r] = (_Float16)(acc[r] + bptr[r]);
            *(v8h*)(dstRow + j * 16 + 8 * hi) = hv;
        }
    }

    // ---- V: original orientation, contiguous v^T row stores ----
#pragma unroll
    for (int j = 0; j < 4; ++j) {
        const _Float16* wrow = WvH + (size_t)(h * D_ + j * 16) * D_;
        v16h wb0 = load_frag<false>(wrow, D_, lane, 0);
        v16h wb1 = load_frag<false>(wrow, D_, lane, 32);
        v8f acc = {};
        acc = wmma_f16(xa0, wb0, acc);
        acc = wmma_f16(xa1, wb1, acc);
        float bo = bv[h * D_ + j * 16 + nn];
        _Float16* dst = vTws + ((size_t)bh * D_ + j * 16 + nn) * T_ + tBase + 8 * hi;
        v8h hv;
#pragma unroll
        for (int r = 0; r < 8; ++r) hv[r] = (_Float16)(acc[r] + bo);
        *(v8h*)dst = hv;
    }
}

// ---------------------------------------------------------------------------
// Flash attention (transposed). Block = 4 waves = 4 consecutive query tiles
// of one (b,h). Each 32-key block of K (32x64) and V^T (64x32) is staged in
// padded LDS via async global->LDS b128 copies and shared by all 4 waves.
// ---------------------------------------------------------------------------
__global__ void flash_attn_kernel(const _Float16* __restrict__ qws,
                                  const _Float16* __restrict__ kws,
                                  const _Float16* __restrict__ vTws,
                                  _Float16* __restrict__ aout) {
    __shared__ _Float16 kLds[32 * KSTR];
    __shared__ _Float16 vLds[64 * VSTR];

    int tid  = threadIdx.x;
    int lane = tid & 31;
    int wv   = tid >> 5;                    // wave in block: 0..3
    const int QG = T_ / 64;                 // query groups per (b,h)
    int qGroup = blockIdx.x % QG;
    int bh     = blockIdx.x / QG;
    int qTile = qGroup * 4 + wv;
    int qBase = qTile * 16;
    int hi = (lane >> 4) & 1;
    int nn = lane & 15;
    int qg = qBase + nn;                    // this lane's query

    const _Float16* qbase = qws + ((size_t)bh * T_ + qBase) * D_;
    v16h qf0 = load_frag<false>(qbase, D_, lane, 0);   // Q^T B-fragments
    v16h qf1 = load_frag<false>(qbase, D_, lane, 32);

    v8f o[4] = {};
    float m = -1e30f, l = 0.f;

    int nb    = (qBase + 16 + 31) / 32;     // this wave's causal block count
    int nbMax = (qGroup + 1) * 2;           // last wave's block count

    const _Float16* kbh = kws  + (size_t)bh * T_ * D_;
    const _Float16* vbh = vTws + (size_t)bh * D_ * T_;

    for (int kb = 0; kb < nbMax; ++kb) {
        int kBase = kb * 32;

        // Cooperative async stage: K tile 32x64 (256 16B chunks),
        // V^T tile 64x32 (256 16B chunks); 128 threads x 2 chunks each.
        const _Float16* kg = kbh + (size_t)kBase * D_;
#pragma unroll
        for (int i = 0; i < 2; ++i) {
            int c = tid + i * 128;
            int kr = c >> 3, ko = c & 7;    // K: 8 chunks per 64-f16 row
            async_copy16((uint32_t)(size_t)&kLds[kr * KSTR + ko * 8],
                         kg + (size_t)kr * D_ + ko * 8);
            int vr = c >> 2, vo = c & 3;    // V^T: 4 chunks per 32-f16 row
            async_copy16((uint32_t)(size_t)&vLds[vr * VSTR + vo * 8],
                         vbh + (size_t)vr * T_ + kBase + vo * 8);
        }
        wait_async0();
        __syncthreads();

        if (kb < nb) {                      // wave-uniform: EXEC stays all-1
            v8f s0 = {}, s1 = {};
            {
                v16h ka0 = load_frag<true>(kLds, KSTR, lane, 0);
                v16h ka1 = load_frag<true>(kLds, KSTR, lane, 32);
                s0 = wmma_f16(ka0, qf0, s0);
                s0 = wmma_f16(ka1, qf1, s0);
                v16h kc0 = load_frag<true>(kLds + 16 * KSTR, KSTR, lane, 0);
                v16h kc1 = load_frag<true>(kLds + 16 * KSTR, KSTR, lane, 32);
                s1 = wmma_f16(kc0, qf0, s1);
                s1 = wmma_f16(kc1, qf1, s1);
            }

            bool maskBlk = (kBase + 31 > qBase);
            float smax = -1e30f;
#pragma unroll
            for (int r = 0; r < 8; ++r) {
                int kg0 = kBase + r + 8 * hi;
                int kg1 = kg0 + 16;
                float v0 = s0[r] * SCALE_;
                float v1 = s1[r] * SCALE_;
                if (maskBlk) {
                    if (kg0 > qg) v0 = -1e30f;
                    if (kg1 > qg) v1 = -1e30f;
                }
                s0[r] = v0; s1[r] = v1;
                smax = fmaxf(smax, fmaxf(v0, v1));
            }
            smax = fmaxf(smax, __shfl_xor(smax, 16, 32));
            float mn    = fmaxf(m, smax);
            float alpha = __expf(m - mn);
            m = mn;

            float e0[8], e1[8];
            float rowsum = 0.f;
#pragma unroll
            for (int r = 0; r < 8; ++r) {
                e0[r] = __expf(s0[r] - mn);
                e1[r] = __expf(s1[r] - mn);
                rowsum += e0[r] + e1[r];
            }
            rowsum += __shfl_xor(rowsum, 16, 32);
            l = l * alpha + rowsum;

            // Build P^T B-fragment: lane needs key K = kBase + 16*hi + e;
            // half resident, half from opposite lane-half (xor-16).
            v16h p;
#pragma unroll
            for (int r = 0; r < 8; ++r) {
                float x0 = __shfl_xor(e0[r], 16, 32);
                float x1 = __shfl_xor(e1[r], 16, 32);
                p[r]     = (_Float16)(hi ? x1 : e0[r]);
                p[8 + r] = (_Float16)(hi ? e1[r] : x0);
            }

            // O^T = alpha*O^T + V^T P^T
#pragma unroll
            for (int dt = 0; dt < 4; ++dt) {
                v16h va = load_frag<true>(vLds + dt * 16 * VSTR, VSTR, lane, 0);
#pragma unroll
                for (int r = 0; r < 8; ++r) o[dt][r] *= alpha;
                o[dt] = wmma_f16(va, p, o[dt]);
            }
        }
        __syncthreads();                    // protect LDS before next stage
    }

    // Epilogue: lane owns one output row; 4 contiguous b128 stores.
    float inv = 1.f / l;
    _Float16* orow = aout + ((size_t)(bh / H_) * T_ + qBase + nn) * E_ +
                     (bh % H_) * D_;
#pragma unroll
    for (int dt = 0; dt < 4; ++dt) {
        v8h hv;
#pragma unroll
        for (int r = 0; r < 8; ++r) hv[r] = (_Float16)(o[dt][r] * inv);
        *(v8h*)(orow + dt * 16 + 8 * hi) = hv;
    }
}

// ---------------------------------------------------------------------------
// Output projection, transposed + 2x2 blocked: out^T = Wp * X^T.
// Wave computes a 32(o) x 32(t) block; lane = t column -> per-lane contiguous
// float stores along o. 4 fragment loads feed 4 WMMAs per K step.
// ---------------------------------------------------------------------------
__global__ void out_proj_kernel(const _Float16* __restrict__ aout,
                                const _Float16* __restrict__ WpH,
                                const float* __restrict__ bp,
                                float* __restrict__ out) {
    int gtid = blockIdx.x * blockDim.x + threadIdx.x;
    int wid  = gtid >> 5;
    int lane = gtid & 31;
    const int OB = E_ / 32;
    int oBlk = wid % OB;
    int tBlk = wid / OB;
    if (tBlk >= (B_ * T_) / 32) return;
    int oBase = oBlk * 32, tBase = tBlk * 32;
    int hi = (lane >> 4) & 1;
    int nn = lane & 15;

    const _Float16* a0 = WpH + (size_t)oBase * E_;          // A: Wp o-rows
    const _Float16* a1 = WpH + (size_t)(oBase + 16) * E_;
    const _Float16* x0 = aout + (size_t)tBase * E_;         // B: X t-rows
    const _Float16* x1 = aout + (size_t)(tBase + 16) * E_;

    v8f acc00 = {}, acc01 = {}, acc10 = {}, acc11 = {};
    for (int kk = 0; kk < E_; kk += 32) {
        v16h A0 = load_frag<true >(a0, E_, lane, kk);
        v16h A1 = load_frag<true >(a1, E_, lane, kk);
        v16h B0 = load_frag<false>(x0, E_, lane, kk);
        v16h B1 = load_frag<false>(x1, E_, lane, kk);
        acc00 = wmma_f16(A0, B0, acc00);
        acc01 = wmma_f16(A0, B1, acc01);
        acc10 = wmma_f16(A1, B0, acc10);
        acc11 = wmma_f16(A1, B1, acc11);
    }

    v8f* accs[4] = {&acc00, &acc01, &acc10, &acc11};
#pragma unroll
    for (int oi = 0; oi < 2; ++oi)
#pragma unroll
        for (int ti = 0; ti < 2; ++ti) {
            v8f a = *accs[oi * 2 + ti];
            int oo = oBase + oi * 16 + 8 * hi;
            int t  = tBase + ti * 16 + nn;
            const float* bptr = bp + oo;
            v4f lo, hip4;
#pragma unroll
            for (int r = 0; r < 4; ++r) {
                lo[r]   = a[r]     + bptr[r];
                hip4[r] = a[r + 4] + bptr[r + 4];
            }
            float* dst = out + (size_t)t * E_ + oo;
            *(v4f*)dst       = lo;
            *(v4f*)(dst + 4) = hip4;
        }
}

// ---------------------------------------------------------------------------
extern "C" void kernel_launch(void* const* d_in, const int* in_sizes, int n_in,
                              void* d_out, int out_size, void* d_ws, size_t ws_size,
                              hipStream_t stream) {
    (void)in_sizes; (void)n_in; (void)out_size; (void)ws_size;

    const float* x  = (const float*)d_in[0];
    // d_in[1] = additive causal mask -> recomputed in-kernel
    const float* Wq = (const float*)d_in[2];
    const float* bq = (const float*)d_in[3];
    const float* Wk = (const float*)d_in[4];
    const float* bk = (const float*)d_in[5];
    const float* Wv = (const float*)d_in[6];
    const float* bv = (const float*)d_in[7];
    const float* Wp = (const float*)d_in[8];
    const float* bp = (const float*)d_in[9];
    float* out = (float*)d_out;

    char*  ws  = (char*)d_ws;
    size_t off = 0;
    auto alloc = [&](size_t bytes) -> char* {
        char* p = ws + off;
        off += (bytes + 255) & ~(size_t)255;
        return p;
    };
    const size_t qkvBytes = (size_t)B_ * H_ * T_ * D_ * sizeof(_Float16);
    const size_t wBytes   = (size_t)H_ * D_ * D_ * sizeof(_Float16);
    _Float16* qws  = (_Float16*)alloc(qkvBytes);
    _Float16* kws  = (_Float16*)alloc(qkvBytes);
    _Float16* vTws = (_Float16*)alloc(qkvBytes);
    _Float16* aout = (_Float16*)alloc((size_t)B_ * T_ * E_ * sizeof(_Float16));
    _Float16* WpH  = (_Float16*)alloc((size_t)E_ * E_ * sizeof(_Float16));
    _Float16* xH   = (_Float16*)alloc((size_t)B_ * T_ * E_ * sizeof(_Float16));
    _Float16* WqH  = (_Float16*)alloc(wBytes);
    _Float16* WkH  = (_Float16*)alloc(wBytes);
    _Float16* WvH  = (_Float16*)alloc(wBytes);

    {   // 1) f32 -> f16 pre-conversion of all WMMA operands
        int n;
        n = E_ * E_;
        cvt_f16_kernel<<<(n + 255) / 256, 256, 0, stream>>>(Wp, WpH, n);
        n = B_ * T_ * E_;
        cvt_f16_kernel<<<(n + 255) / 256, 256, 0, stream>>>(x, xH, n);
        n = H_ * D_ * D_;
        cvt_f16_kernel<<<(n + 255) / 256, 256, 0, stream>>>(Wq, WqH, n);
        cvt_f16_kernel<<<(n + 255) / 256, 256, 0, stream>>>(Wk, WkH, n);
        cvt_f16_kernel<<<(n + 255) / 256, 256, 0, stream>>>(Wv, WvH, n);
    }
    {   // 2) QKV projection: 4096 waves
        int waves = B_ * H_ * (T_ / 16);
        qkv_proj_kernel<<<waves / 4, 128, 0, stream>>>(xH, WqH, bq, WkH, bk,
                                                       WvH, bv, qws, kws, vTws);
    }
    {   // 3) Flash attention: one block (4 waves) per (b,h,64-query group)
        int blocks = B_ * H_ * (T_ / 64);
        flash_attn_kernel<<<blocks, 128, 0, stream>>>(qws, kws, vTws, aout);
    }
    {   // 4) Output projection: 4096 waves (32x32 tiles)
        int waves = (B_ * T_ / 32) * (E_ / 32);
        out_proj_kernel<<<waves / 4, 128, 0, stream>>>(aout, WpH, bp, out);
    }
}
